// VideoSwinTransformer_10763188043924
// MI455X (gfx1250) — compile-verified, matrix operand order checked by reading, not run on hardware
//
#include <hip/hip_runtime.h>
#include <hip/hip_bf16.h>
#include <math.h>

typedef __attribute__((ext_vector_type(16))) _Float16     v16h;
typedef __attribute__((ext_vector_type(8)))  float        v8f;
typedef __attribute__((ext_vector_type(4)))  unsigned int v4u;
typedef __attribute__((ext_vector_type(8)))  int          v8i;
typedef __attribute__((ext_vector_type(4)))  int          v4i;

#define DIM   192
#define NHEAD 6
#define HD    32
#define NTOK  200704          // 64 * 3136
#define MLPH  768
#define KC    192             // K-chunk staged into LDS per TDM transfer
#define NBLK  64              // N columns per block (B panel rows)
#define MBLK  256             // M rows per block (8 waves x 32 rows)

// ---------------------------------------------------------------------------
// WMMA fragment loads.  16-bit A-matrix 16x32 layout (ISA 7.12.2):
//   lanes 0-15 : row = lane,    K = {0..7, 16..23}
//   lanes16-31 : row = lane-16, K = {8..15, 24..31}
// B operand uses the identical per-lane pattern over rows of W (= cols of W^T).
// ---------------------------------------------------------------------------
__device__ __forceinline__ v16h load_frag32(const _Float16* __restrict__ p, int lane) {
    const int off = (lane & 16) ? 8 : 0;
    v16h r;
#pragma unroll
    for (int i = 0; i < 8; ++i)  r[i] = p[off + i];
#pragma unroll
    for (int i = 8; i < 16; ++i) r[i] = p[off + i + 8];
    return r;
}

__device__ __forceinline__ v8f wmma_f16(v16h a, v16h b, v8f c) {
    return __builtin_amdgcn_wmma_f32_16x16x32_f16(false, a, false, b, (short)0, c, false, false);
}

// LDS byte offset of a generic pointer to shared memory (for the TDM D#).
__device__ __forceinline__ unsigned lds_offset(const void* p) {
    return (unsigned)(unsigned long long)
        (const __attribute__((address_space(3))) char*)(const char*)p;
}

// Toolchain-portable TDM issue (5-arg ROCm 7.2 clang-22 vs 6-arg therock).
#if __has_include(<hip/amd_detail/amd_gfx1250_TDM.h>)
#define TDM_LOAD_TO_LDS(g0, g1, z4a, z4b, z8) \
    __builtin_amdgcn_tensor_load_to_lds((g0), (g1), (z4a), (z4b), (z8), 0)
#else
#define TDM_LOAD_TO_LDS(g0, g1, z4a, z4b, z8) \
    __builtin_amdgcn_tensor_load_to_lds((g0), (g1), (z4a), (z4b), 0)
#endif

// Issue a 2D TDM load: tile = 64 rows x KC cols of W starting at `gaddr`,
// row stride = K elements (2B each), into LDS at byte offset `ldsoff`.
__device__ __forceinline__ void tdm_load_panel(unsigned long long gaddr,
                                               unsigned ldsoff, int K) {
    v4u g0 = { 1u,                                   // count=1, user mode
               ldsoff,                               // lds_addr (bytes)
               (unsigned)(gaddr & 0xFFFFFFFFu),
               (unsigned)(((gaddr >> 32) & 0x1FFFFFFu) | (2u << 30)) }; // type=2
    v8i g1 = { (int)(1u << 16),                      // data_size=1 (2 bytes)
               (int)(((unsigned)KC & 0xFFFFu) << 16),// tensor_dim0 lo16
               (int)(64u << 16),                     // tensor_dim0 hi | tensor_dim1 lo = 64
               (int)((unsigned)KC << 16),            // tensor_dim1 hi | tile_dim0 = KC
               (int)64,                              // tile_dim1 = 64, tile_dim2 = 0
               (int)K,                               // tensor_dim0_stride lo32 = K
               0, 0 };                               // stride hi / dim1 stride = 0
    v4i z4 = {0, 0, 0, 0};
    v8i z8 = {0, 0, 0, 0, 0, 0, 0, 0};
    TDM_LOAD_TO_LDS(g0, g1, z4, z4, z8);
}

// ---------------------------------------------------------------------------
// fp32 -> fp16 weight conversion
// ---------------------------------------------------------------------------
__global__ void f2h_kernel(const float* __restrict__ s, _Float16* __restrict__ d, int n) {
    int i = blockIdx.x * 256 + threadIdx.x;
    if (i < n) d[i] = (_Float16)s[i];
}

// ---------------------------------------------------------------------------
// LayerNorm, wave-per-token.  WIN=true additionally applies the -3 cyclic
// shift + 7x7 window partition (gather): output token order is windowed.
// ---------------------------------------------------------------------------
template<bool WIN>
__global__ __launch_bounds__(256) void ln_kernel(const float* __restrict__ src,
                                                 const float* __restrict__ g,
                                                 const float* __restrict__ b,
                                                 _Float16* __restrict__ dst, int ntok) {
    int lane = threadIdx.x & 31;
    int t = blockIdx.x * 8 + (threadIdx.x >> 5);
    if (t >= ntok) return;
    int p;
    if (WIN) {
        int bb = t / 3136, rem = t % 3136;
        int w = rem / 49, nI = rem % 49;
        int wy = w >> 3, wx = w & 7;
        int iy = nI / 7, ix = nI % 7;
        int y = (wy * 7 + iy + 3) % 56;   // roll(-3): src = pos + 3 (mod 56)
        int x = (wx * 7 + ix + 3) % 56;
        p = bb * 3136 + y * 56 + x;
    } else {
        p = t;
    }
    const float* xp = src + (size_t)p * DIM;
    float v[6]; float s = 0.f;
#pragma unroll
    for (int i = 0; i < 6; ++i) { v[i] = xp[i * 32 + lane]; s += v[i]; }
#pragma unroll
    for (int o = 16; o >= 1; o >>= 1) s += __shfl_xor(s, o, 32);
    float mu = s * (1.f / 192.f);
    float q = 0.f;
#pragma unroll
    for (int i = 0; i < 6; ++i) { float d = v[i] - mu; q += d * d; }
#pragma unroll
    for (int o = 16; o >= 1; o >>= 1) q += __shfl_xor(q, o, 32);
    float rstd = rsqrtf(q * (1.f / 192.f) + 1e-6f);
    _Float16* dp = dst + (size_t)t * DIM;
#pragma unroll
    for (int i = 0; i < 6; ++i) {
        int ch = i * 32 + lane;
        dp[ch] = (_Float16)((v[i] - mu) * rstd * g[ch] + b[ch]);
    }
}

// ---------------------------------------------------------------------------
// GEMM: out = A(MxK,f16) @ W(NxK,f16)^T + bias.
// Block = 256 threads (8 waves), block tile 256(M) x 64(N).
// Wave tile 32x64: 2 A-frags x 4 B-frags -> 8 wmma per 32-wide k-step.
// W panel (64 rows x KC cols) staged into LDS by the Tensor Data Mover,
// synchronized with s_wait_tensorcnt + barrier; A streams from global (L2).
// MODE 0: store f16                     (QKV)
// MODE 1: exact GELU, store f16         (fc1)
// MODE 2: window-reverse scatter + residual(x) -> f32 out   (proj)
// MODE 3: in-place residual add on f32 out                  (fc2)
// ---------------------------------------------------------------------------
template<int MODE>
__global__ __launch_bounds__(256) void gemm_x_wT(const _Float16* __restrict__ A,
                                                 const _Float16* __restrict__ W,
                                                 const float* __restrict__ bias,
                                                 _Float16* __restrict__ outH,
                                                 float* __restrict__ outF,
                                                 const float* __restrict__ xres,
                                                 int M, int N, int K) {
    extern __shared__ _Float16 Bs[];          // 64 x KC panel of W
    int lane = threadIdx.x & 31;
    int wave = threadIdx.x >> 5;
    int n64 = N >> 6;
    int mblk = blockIdx.x / n64;
    int nblk = blockIdx.x % n64;
    int nbase = nblk << 6;
    int m0 = mblk * MBLK + wave * 32;
    if (m0 >= M) return;

    const _Float16* ap0 = A + (size_t)(m0 + (lane & 15)) * K;
    const _Float16* ap1 = ap0 + (size_t)16 * K;

    v8f zero = {};
    v8f acc[8];
#pragma unroll
    for (int s = 0; s < 8; ++s) acc[s] = zero;

    for (int kc = 0; kc < K; kc += KC) {
        __syncthreads();                       // previous panel fully consumed
        if (wave == 0) {
            unsigned long long ga =
                (unsigned long long)(const void*)(W + (size_t)nbase * K + kc);
            tdm_load_panel(ga, lds_offset(Bs), K);
            __builtin_amdgcn_s_wait_tensorcnt(0);
        }
        __syncthreads();                       // panel visible to all waves

        if (kc + KC < K)                       // prefetch next A chunk
            __builtin_prefetch(ap0 + kc + KC, 0, 1);

#pragma unroll
        for (int kb = 0; kb < KC; kb += 32) {
            v16h a0 = load_frag32(ap0 + kc + kb, lane);
            v16h a1 = load_frag32(ap1 + kc + kb, lane);
#pragma unroll
            for (int s = 0; s < 4; ++s) {
                const _Float16* bp = Bs + (size_t)(s * 16 + (lane & 15)) * KC + kb;
                v16h bf = load_frag32(bp, lane);
                acc[s]     = wmma_f16(a0, bf, acc[s]);
                acc[4 + s] = wmma_f16(a1, bf, acc[4 + s]);
            }
        }
    }

#pragma unroll
    for (int mh = 0; mh < 2; ++mh) {
        int rbase = m0 + mh * 16 + ((lane & 16) ? 8 : 0);
#pragma unroll
        for (int s = 0; s < 4; ++s) {
            int col = nbase + s * 16 + (lane & 15);
            float bv = bias[col];
#pragma unroll
            for (int j = 0; j < 8; ++j) {
                int r = rbase + j;
                float v = acc[mh * 4 + s][j] + bv;
                if (MODE == 0) {
                    outH[(size_t)r * N + col] = (_Float16)v;
                } else if (MODE == 1) {
                    float ge = 0.5f * v * (1.0f + erff(v * 0.70710678118654752f));
                    outH[(size_t)r * N + col] = (_Float16)ge;
                } else if (MODE == 2) {
                    int bb = r / 3136, rem = r % 3136;
                    int w = rem / 49, nI = rem % 49;
                    int wy = w >> 3, wx = w & 7;
                    int iy = nI / 7, ix = nI % 7;
                    int y = (wy * 7 + iy + 3) % 56;   // roll(+3): dst = pos + 3
                    int x = (wx * 7 + ix + 3) % 56;
                    size_t dst = ((size_t)bb * 3136 + y * 56 + x) * DIM + col;
                    outF[dst] = v + xres[dst];
                } else {  // MODE 3
                    size_t idx = (size_t)r * DIM + col;
                    outF[idx] = outF[idx] + v;
                }
            }
        }
    }
}

// ---------------------------------------------------------------------------
// Windowed attention.  One block = (window, head), 4 waves = 4 row-tiles.
// qkv layout: (NTOK, 576) f16, cols [0,192)=Q [192,384)=K [384,576)=V.
// ---------------------------------------------------------------------------
__global__ __launch_bounds__(128) void attn_kernel(const _Float16* __restrict__ qkv,
                                                   const float* __restrict__ rel_bias,
                                                   _Float16* __restrict__ outH) {
    __shared__ float    S[64][68];
    __shared__ _Float16 P[64][72];
    __shared__ _Float16 VT[32][72];

    int w = blockIdx.x;          // 0..4095 (b*64 + window)
    int head = blockIdx.y;       // 0..5
    int tid = threadIdx.x;
    int lane = tid & 31, wave = tid >> 5;
    int wi = w & 63, wy = wi >> 3, wx = wi & 7;

    const _Float16* base = qkv + (size_t)w * 49 * 576;
    const _Float16* Qp = base + head * HD;
    const _Float16* Kp = base + DIM + head * HD;
    const _Float16* Vp = base + 2 * DIM + head * HD;

    // stage V transposed (HD x 64), zero-padded tokens
    for (int idx = tid; idx < 64 * 32; idx += 128) {
        int r = idx >> 5, c = idx & 31;
        VT[c][r] = (r < 49) ? Vp[(size_t)r * 576 + c] : (_Float16)0.f;
    }

    // ---- S = Q @ K^T (K-dim = head dim = 32, single wmma per tile pair) ----
    int m0 = wave * 16;
    int qr = m0 + (lane & 15); if (qr > 48) qr = 48;
    v16h af = load_frag32(Qp + (size_t)qr * 576, lane);
    v8f zero = {};
    v8f s4[4];
#pragma unroll
    for (int n = 0; n < 4; ++n) {
        int kr = n * 16 + (lane & 15); if (kr > 48) kr = 48;
        v16h bf = load_frag32(Kp + (size_t)kr * 576, lane);
        s4[n] = wmma_f16(af, bf, zero);
    }

    const float scale = 0.17677669529663687f;   // 32^-0.5
    int rbase = m0 + ((lane & 16) ? 8 : 0);
#pragma unroll
    for (int n = 0; n < 4; ++n) {
#pragma unroll
        for (int j = 0; j < 8; ++j) {
            int row = rbase + j;
            int col = n * 16 + (lane & 15);
            float val;
            if (col < 49 && row < 49) {
                int ry = row / 7, rx = row % 7;
                int cy = col / 7, cx = col % 7;
                int relidx = (ry - cy + 6) * 13 + (rx - cx + 6);
                float bv = rel_bias[relidx * NHEAD + head];
                int ygr = wy * 7 + ry, xgr = wx * 7 + rx;
                int ygc = wy * 7 + cy, xgc = wx * 7 + cx;
                int idr = (ygr < 49 ? 0 : (ygr < 53 ? 1 : 2)) * 3 +
                          (xgr < 49 ? 0 : (xgr < 53 ? 1 : 2));
                int idc = (ygc < 49 ? 0 : (ygc < 53 ? 1 : 2)) * 3 +
                          (xgc < 49 ? 0 : (xgc < 53 ? 1 : 2));
                val = s4[n][j] * scale + bv + (idr == idc ? 0.f : -100.f);
            } else {
                val = -1e30f;   // pad -> exp()=0
            }
            S[row][col] = val;
        }
    }
    __syncthreads();

    // ---- softmax rows (thread-per-row), clamp at 1e-8 as in reference ----
    if (tid < 64) {
        int row = tid;
        if (row < 49) {
            float mx = -1e30f;
            for (int c = 0; c < 49; ++c) mx = fmaxf(mx, S[row][c]);
            float sum = 0.f;
            for (int c = 0; c < 49; ++c) {
                float e = __expf(S[row][c] - mx);
                S[row][c] = e;
                sum += e;
            }
            float inv = 1.f / sum;
            for (int c = 0; c < 49; ++c) P[row][c] = (_Float16)fmaxf(S[row][c] * inv, 1e-8f);
            for (int c = 49; c < 64; ++c) P[row][c] = (_Float16)0.f;
        } else {
            for (int c = 0; c < 64; ++c) P[row][c] = (_Float16)0.f;
        }
    }
    __syncthreads();

    // ---- O = P @ V : K-dim = 64 tokens (2 k-steps), N = 32 (2 tiles) ----
    v16h pa[2];
#pragma unroll
    for (int kb = 0; kb < 2; ++kb)
        pa[kb] = load_frag32(&P[m0 + (lane & 15)][kb * 32], lane);

#pragma unroll
    for (int n = 0; n < 2; ++n) {
        v8f o = zero;
#pragma unroll
        for (int kb = 0; kb < 2; ++kb) {
            v16h bf = load_frag32(&VT[n * 16 + (lane & 15)][kb * 32], lane);
            o = wmma_f16(pa[kb], bf, o);
        }
#pragma unroll
        for (int j = 0; j < 8; ++j) {
            int row = rbase + j;
            if (row < 49) {
                int col = head * HD + n * 16 + (lane & 15);
                outH[((size_t)w * 49 + row) * DIM + col] = (_Float16)o[j];
            }
        }
    }
}

// ---------------------------------------------------------------------------
// Host launcher
// ---------------------------------------------------------------------------
static inline int gemm_blocks(int M, int N) {
    long tiles = (long)((M + MBLK - 1) / MBLK) * (N / NBLK);
    return (int)tiles;
}

extern "C" void kernel_launch(void* const* d_in, const int* in_sizes, int n_in,
                              void* d_out, int out_size, void* d_ws, size_t ws_size,
                              hipStream_t stream) {
    const float* x      = (const float*)d_in[0];
    const float* qkv_w  = (const float*)d_in[1];
    const float* qkv_b  = (const float*)d_in[2];
    const float* proj_w = (const float*)d_in[3];
    const float* proj_b = (const float*)d_in[4];
    const float* relb   = (const float*)d_in[5];
    const float* n1g    = (const float*)d_in[6];
    const float* n1b    = (const float*)d_in[7];
    const float* n2g    = (const float*)d_in[8];
    const float* n2b    = (const float*)d_in[9];
    const float* fc1_w  = (const float*)d_in[10];
    const float* fc1_b  = (const float*)d_in[11];
    const float* fc2_w  = (const float*)d_in[12];
    const float* fc2_b  = (const float*)d_in[13];
    float* out = (float*)d_out;

    char* ws = (char*)d_ws;
    size_t off = 0;
    auto alloc = [&](size_t bytes) -> char* {
        char* p = ws + off;
        off += (bytes + 255) & ~(size_t)255;
        return p;
    };
    _Float16* wq  = (_Float16*)alloc((size_t)3 * DIM * DIM * 2);
    _Float16* wp  = (_Float16*)alloc((size_t)DIM * DIM * 2);
    _Float16* w1  = (_Float16*)alloc((size_t)MLPH * DIM * 2);
    _Float16* w2  = (_Float16*)alloc((size_t)DIM * MLPH * 2);
    _Float16* xw  = (_Float16*)alloc((size_t)NTOK * DIM * 2);    // LN1 out; reused as attn out
    _Float16* qkv = (_Float16*)alloc((size_t)NTOK * 3 * DIM * 2);// qkv; reused as LN2 out
    _Float16* f1h = (_Float16*)alloc((size_t)NTOK * MLPH * 2);   // fc1 out
    _Float16* attn = xw;
    _Float16* h2   = qkv;

    // weight conversion fp32 -> fp16
    {
        int n0 = 3 * DIM * DIM, n1 = DIM * DIM, n2 = MLPH * DIM, n3 = DIM * MLPH;
        f2h_kernel<<<(n0 + 255) / 256, 256, 0, stream>>>(qkv_w, wq, n0);
        f2h_kernel<<<(n1 + 255) / 256, 256, 0, stream>>>(proj_w, wp, n1);
        f2h_kernel<<<(n2 + 255) / 256, 256, 0, stream>>>(fc1_w, w1, n2);
        f2h_kernel<<<(n3 + 255) / 256, 256, 0, stream>>>(fc2_w, w2, n3);
    }

    const int lnBlocks = (NTOK + 7) / 8;
    const size_t ldsB = (size_t)64 * KC * 2;   // 24 KB W panel

    // 1) LN1 + shift + window partition -> xw (windowed token order, f16)
    ln_kernel<true><<<lnBlocks, 256, 0, stream>>>(x, n1g, n1b, xw, NTOK);

    // 2) QKV GEMM: (NTOK,192) @ (576,192)^T -> (NTOK,576)
    gemm_x_wT<0><<<gemm_blocks(NTOK, 3 * DIM), 256, ldsB, stream>>>(
        xw, wq, qkv_b, qkv, nullptr, nullptr, NTOK, 3 * DIM, DIM);

    // 3) windowed attention: 4096 windows x 6 heads
    attn_kernel<<<dim3(4096, NHEAD), 128, 0, stream>>>(qkv, relb, attn);

    // 4) proj GEMM + window-reverse + un-shift + residual(x) -> d_out (f32)
    gemm_x_wT<2><<<gemm_blocks(NTOK, DIM), 256, ldsB, stream>>>(
        attn, wp, proj_b, nullptr, out, x, NTOK, DIM, DIM);

    // 5) LN2 on d_out -> h2 (f16)
    ln_kernel<false><<<lnBlocks, 256, 0, stream>>>(out, n2g, n2b, h2, NTOK);

    // 6) fc1 GEMM + exact GELU -> f1h
    gemm_x_wT<1><<<gemm_blocks(NTOK, MLPH), 256, ldsB, stream>>>(
        h2, w1, fc1_b, f1h, nullptr, nullptr, NTOK, MLPH, DIM);

    // 7) fc2 GEMM (K=768, staged in 4 panels) + residual add in place on d_out
    gemm_x_wT<3><<<gemm_blocks(NTOK, DIM), 256, ldsB, stream>>>(
        f1h, w2, fc2_b, nullptr, out, nullptr, NTOK, DIM, MLPH);
}
